// EarthSpecificLayer_65034394796758
// MI455X (gfx1250) — compile-verified, matrix operand order checked by reading, not run on hardware
//
#include <hip/hip_runtime.h>

typedef __attribute__((ext_vector_type(16))) _Float16 v16h;
typedef __attribute__((ext_vector_type(8)))  float    v8f;

#define WMMA_F32_F16(a, b, c) \
  __builtin_amdgcn_wmma_f32_16x16x32_f16(false, (a), false, (b), (short)0, (c), false, false)

// Load a 16x32 f16 fragment (A layout; B uses the same pattern applied to the
// transposed matrix) from LDS. Lane r in [0,16) -> row r, halves {0..7,16..23};
// lanes 16..31 -> same rows, halves {8..15,24..31}.  Two contiguous 16B reads.
__device__ __forceinline__ v16h frag_ld(const _Float16* __restrict__ base,
                                        int strideH, int lane) {
  const int r  = lane & 15;
  const int hi = (lane >> 4) & 1;
  const _Float16* p = base + r * strideH + hi * 8;
  v16h f;
#pragma unroll
  for (int i = 0; i < 8; ++i) { f[i] = p[i]; f[8 + i] = p[16 + i]; }
  return f;
}

// ---------------------------------------------------------------------------
// Kernel 1: qkv = (x @ w1 + b1), q pre-scaled by Ch^-0.5, stored f16.
// x: 138240x192 f32, w1: 192x576 f32, qkv: 138240x576 f16.
// Block: 64 rows x 64 cols, 8 waves (4 row strips x 2 col halves), K chunks of 32.
// ---------------------------------------------------------------------------
__global__ __launch_bounds__(256) void qkv_gemm(const float* __restrict__ x,
                                                const float* __restrict__ w1,
                                                const float* __restrict__ b1,
                                                _Float16* __restrict__ qkv) {
  __shared__ _Float16 Xs[64 * 200];  // 64 rows x K=192 (stride 200)
  __shared__ _Float16 Wt[64 * 200];  // 64 out-cols x K=192 (stride 200)
  const int m0 = blockIdx.x * 64;
  const int c0 = blockIdx.y * 64;
  const int tid = threadIdx.x;

  const float4* x4 = (const float4*)(x + (size_t)m0 * 192);
  for (int e = tid; e < 64 * 48; e += 256) {
    int r = e / 48, c4 = e % 48;
    float4 v = x4[r * 48 + c4];
    _Float16* d = &Xs[r * 200 + 4 * c4];
    d[0] = (_Float16)v.x; d[1] = (_Float16)v.y;
    d[2] = (_Float16)v.z; d[3] = (_Float16)v.w;
  }
  for (int e = tid; e < 64 * 192; e += 256) {
    int n = e / 192, k = e % 192;
    Wt[n * 200 + k] = (_Float16)w1[(size_t)k * 576 + c0 + n];
  }
  __syncthreads();

  const int lane = tid & 31;
  const int wv   = tid >> 5;
  const int strip = wv & 3;   // which 16-row strip
  const int ch2   = wv >> 2;  // which 32-col half
  v8f acc[2] = {{0,0,0,0,0,0,0,0}, {0,0,0,0,0,0,0,0}};
#pragma unroll
  for (int kc = 0; kc < 6; ++kc) {
    v16h a = frag_ld(&Xs[(16 * strip) * 200 + 32 * kc], 200, lane);
#pragma unroll
    for (int t = 0; t < 2; ++t) {
      v16h b = frag_ld(&Wt[(32 * ch2 + 16 * t) * 200 + 32 * kc], 200, lane);
      acc[t] = WMMA_F32_F16(a, b, acc[t]);
    }
  }
  const float qscale = 0.17677669529663687f;  // 32^-0.5 folded into q
  const int r = lane & 15, hi = lane >> 4;
#pragma unroll
  for (int t = 0; t < 2; ++t) {
    int col = c0 + 32 * ch2 + 16 * t + r;
    float bias = b1[col];
    float s = (col < 192) ? qscale : 1.0f;
#pragma unroll
    for (int j = 0; j < 8; ++j) {
      int row = m0 + 16 * strip + j + 8 * hi;
      qkv[(size_t)row * 576 + col] = (_Float16)((acc[t][j] + bias) * s);
    }
  }
}

// ---------------------------------------------------------------------------
// Kernel 2: per-(window, head) attention. 9 waves; wave w owns rows 16w..16w+15.
// S = Q K^T (+bias+mask), in-register softmax via 16-lane xor shuffles,
// O = P V with K padded 144->160.
// ---------------------------------------------------------------------------
__global__ __launch_bounds__(288) void win_attn(
    const _Float16* __restrict__ qkv, const float* __restrict__ bias_table,
    const float* __restrict__ mask, const int* __restrict__ pos_idx,
    _Float16* __restrict__ attn_out) {
  __shared__ _Float16 U[23040];     // [Qs 4608 | Ks 4608] then reused as Pb[144*160]
  __shared__ _Float16 Vt[32 * 160]; // V transposed, k padded to 160
  _Float16* Qs = U;
  _Float16* Ks = U + 4608;
  _Float16* Pb = U;

  const int win = blockIdx.x / 6;
  const int h   = blockIdx.x % 6;
  const int tid = threadIdx.x;
  const int nWi = win / 15;  // window index within (z,h) pair dim; wlon = 15

  // warm the per-window mask slice (144*144*4B) -> global_prefetch_b8
  __builtin_prefetch(mask + (size_t)win * 20736 + tid * 72, 0, 1);

  const _Float16* base = qkv + (size_t)win * 144 * 576 + h * 32;
  for (int e = tid; e < 4608; e += 288) {
    int n = e >> 5, c = e & 31;
    Qs[e]           = base[n * 576 + c];
    Ks[e]           = base[n * 576 + 192 + c];
    Vt[c * 160 + n] = base[n * 576 + 384 + c];
  }
  for (int e = tid; e < 512; e += 288) {  // zero the k=144..159 pad of V^T
    int c = e >> 4, n = 144 + (e & 15);
    Vt[c * 160 + n] = (_Float16)0.0f;
  }
  __syncthreads();

  const int lane = tid & 31;
  const int wv = tid >> 5;
  const int r = lane & 15, hi = lane >> 4;

  v16h aq = frag_ld(&Qs[(16 * wv) * 32], 32, lane);
  v8f s[9];
#pragma unroll
  for (int t = 0; t < 9; ++t) {
    v16h bk = frag_ld(&Ks[(16 * t) * 32], 32, lane);
    v8f z = {0,0,0,0,0,0,0,0};
    s[t] = WMMA_F32_F16(aq, bk, z);  // full K=32 in one wmma
  }
  // bias gather + shift-mask add.  C layout: vgpr j, lane -> row 16w+j+8*hi,
  // col 16t + (lane&15).
#pragma unroll
  for (int j = 0; j < 8; ++j) {
    int i = 16 * wv + j + 8 * hi;
    const int*   pi   = pos_idx + i * 144;
    const float* mrow = mask + ((size_t)win * 144 + i) * 144;
#pragma unroll
    for (int t = 0; t < 9; ++t) {
      int jc = 16 * t + r;
      int p = pi[jc];
      s[t][j] += bias_table[((size_t)p * 64 + nWi) * 6 + h] + mrow[jc];
    }
  }
  // row softmax: each row's 144 values live in one vgpr slot across 16 lanes
  // x 9 tiles; reduce locally over tiles then xor-shuffle over the 16-lane half.
#pragma unroll
  for (int j = 0; j < 8; ++j) {
    float m = -1e30f;
#pragma unroll
    for (int t = 0; t < 9; ++t) m = fmaxf(m, s[t][j]);
#pragma unroll
    for (int d = 1; d < 16; d <<= 1) m = fmaxf(m, __shfl_xor(m, d, 32));
    float sum = 0.0f;
#pragma unroll
    for (int t = 0; t < 9; ++t) {
      float e = __expf(s[t][j] - m);
      s[t][j] = e;
      sum += e;
    }
#pragma unroll
    for (int d = 1; d < 16; d <<= 1) sum += __shfl_xor(sum, d, 32);
    float inv = 1.0f / sum;
#pragma unroll
    for (int t = 0; t < 9; ++t) s[t][j] *= inv;
  }
  __syncthreads();  // everyone done reading Qs/Ks before Pb overwrites them
#pragma unroll
  for (int j = 0; j < 8; ++j) {
    int row = 16 * wv + j + 8 * hi;
#pragma unroll
    for (int t = 0; t < 9; ++t) Pb[row * 160 + 16 * t + r] = (_Float16)s[t][j];
    Pb[row * 160 + 144 + r] = (_Float16)0.0f;  // pad cols 144..159
  }
  __syncthreads();
  // O(16x32) = P(16x160) @ Vpad(160x32): 2 col tiles x 5 K-chunks of 32
#pragma unroll
  for (int ct = 0; ct < 2; ++ct) {
    v8f o = {0,0,0,0,0,0,0,0};
#pragma unroll
    for (int kc = 0; kc < 5; ++kc) {
      v16h ap = frag_ld(&Pb[(16 * wv) * 160 + 32 * kc], 160, lane);
      v16h bv = frag_ld(&Vt[(16 * ct) * 160 + 32 * kc], 160, lane);
      o = WMMA_F32_F16(ap, bv, o);
    }
#pragma unroll
    for (int j = 0; j < 8; ++j) {
      int n = 16 * wv + j + 8 * hi;
      int c = 16 * ct + r;
      attn_out[((size_t)win * 144 + n) * 192 + h * 32 + c] = (_Float16)o[j];
    }
  }
}

// ---------------------------------------------------------------------------
// Kernel 3: out = attn_out(f16) @ w2 + b2, f32 output.
// ---------------------------------------------------------------------------
__global__ __launch_bounds__(256) void out_gemm(const _Float16* __restrict__ a,
                                                const float* __restrict__ w2,
                                                const float* __restrict__ b2,
                                                float* __restrict__ out) {
  __shared__ _Float16 As[64 * 200];
  __shared__ _Float16 Wt[64 * 200];
  const int m0 = blockIdx.x * 64;
  const int c0 = blockIdx.y * 64;
  const int tid = threadIdx.x;

  const uint32_t* a32 = (const uint32_t*)(a + (size_t)m0 * 192);
  for (int e = tid; e < 64 * 96; e += 256) {  // copy f16 pairs
    int row = e / 96, c2 = e % 96;
    *(uint32_t*)&As[row * 200 + 2 * c2] = a32[row * 96 + c2];
  }
  for (int e = tid; e < 64 * 192; e += 256) {
    int n = e / 192, k = e % 192;
    Wt[n * 200 + k] = (_Float16)w2[(size_t)k * 192 + c0 + n];
  }
  __syncthreads();

  const int lane = tid & 31;
  const int wv = tid >> 5;
  const int strip = wv & 3;
  const int ch2 = wv >> 2;
  v8f acc[2] = {{0,0,0,0,0,0,0,0}, {0,0,0,0,0,0,0,0}};
#pragma unroll
  for (int kc = 0; kc < 6; ++kc) {
    v16h av = frag_ld(&As[(16 * strip) * 200 + 32 * kc], 200, lane);
#pragma unroll
    for (int t = 0; t < 2; ++t) {
      v16h b = frag_ld(&Wt[(32 * ch2 + 16 * t) * 200 + 32 * kc], 200, lane);
      acc[t] = WMMA_F32_F16(av, b, acc[t]);
    }
  }
  const int r = lane & 15, hi = lane >> 4;
#pragma unroll
  for (int t = 0; t < 2; ++t) {
    int col = c0 + 32 * ch2 + 16 * t + r;
    float bias = b2[col];
#pragma unroll
    for (int j = 0; j < 8; ++j) {
      int row = m0 + 16 * strip + j + 8 * hi;
      out[(size_t)row * 192 + col] = acc[t][j] + bias;
    }
  }
}

extern "C" void kernel_launch(void* const* d_in, const int* in_sizes, int n_in,
                              void* d_out, int out_size, void* d_ws,
                              size_t ws_size, hipStream_t stream) {
  const float* x          = (const float*)d_in[0];
  const float* w1         = (const float*)d_in[1];
  const float* b1         = (const float*)d_in[2];
  const float* w2         = (const float*)d_in[3];
  const float* b2         = (const float*)d_in[4];
  const float* bias_table = (const float*)d_in[5];
  const float* mask       = (const float*)d_in[6];
  const int*   pos        = (const int*)d_in[7];
  float* out = (float*)d_out;

  // Workspace: qkv f16 (138240x576 = 159.3 MB) | attn f16 (138240x192 = 53.1 MB)
  _Float16* qkv = (_Float16*)d_ws;
  _Float16* att = qkv + (size_t)138240 * 576;

  qkv_gemm<<<dim3(2160, 9), 256, 0, stream>>>(x, w1, b1, qkv);
  win_attn<<<dim3(5760), 288, 0, stream>>>(qkv, bias_table, mask, pos, att);
  out_gemm<<<dim3(2160, 3), 256, 0, stream>>>(att, w2, b2, out);
}
